// Attention_84335977824434
// MI455X (gfx1250) — matrix operand order repaired and checked by reading, not compiled, not run on hardware
//
#include <hip/hip_runtime.h>
#include <hip/hip_bf16.h>

// ---- problem constants (from reference) ----
#define B_  32
#define T_  2048
#define D_  1024
#define U_  1024

typedef __attribute__((ext_vector_type(16))) __bf16 v16bf;
typedef __attribute__((ext_vector_type(8)))  float  v8f;
typedef __attribute__((ext_vector_type(4)))  float  v4f;

// ---------------------------------------------------------------------------
// Prep 1: W2 [D,U] f32 (row-major) -> W2T [U,D] bf16, so each B-matrix column
// is K-contiguous (one 32B load per lane per K-step of 32).
// ---------------------------------------------------------------------------
__global__ void w2_transpose_bf16_kernel(const float* __restrict__ W2,
                                         __bf16* __restrict__ w2t) {
    int idx = blockIdx.x * 256 + threadIdx.x;   // 0 .. U_*D_-1
    int u = idx >> 10;                          // / D_
    int k = idx & (D_ - 1);
    w2t[idx] = (__bf16)W2[k * U_ + u];
}

// ---------------------------------------------------------------------------
// Prep 2: qb[b,u] = query[b,:] @ W1[:,u] + b1[u] + b2[u]   (tiny: 33 MFLOP)
// ---------------------------------------------------------------------------
__global__ void qproj_kernel(const float* __restrict__ query,
                             const float* __restrict__ W1,
                             const float* __restrict__ b1,
                             const float* __restrict__ b2,
                             float* __restrict__ qb) {
    int b = blockIdx.x >> 2;                       // U_/256 = 4 chunks per b
    int u = ((blockIdx.x & 3) << 8) + threadIdx.x;
    float acc = b1[u] + b2[u];
    const float* qr = query + b * D_;
    for (int d = 0; d < D_; ++d)
        acc += qr[d] * W1[d * U_ + u];
    qb[b * U_ + u] = acc;
}

__global__ void zero_kernel(float* __restrict__ p) {
    p[blockIdx.x * 256 + threadIdx.x] = 0.0f;
}

// ---------------------------------------------------------------------------
// Load a 16x32 f32 A-tile row-slice for this lane and convert to the bf16
// WMMA A fragment. K order used here (contiguous, same as B): lane<16 holds K={0..15},
// lane>=16 holds K={16..31}; 'ap' is pre-offset by hi*16.
// ---------------------------------------------------------------------------
__device__ __forceinline__ v16bf load_a_frag(const float* __restrict__ ap) {
    v4f x0 = *(const v4f*)(ap);
    v4f x1 = *(const v4f*)(ap + 4);
    v4f x2 = *(const v4f*)(ap + 8);
    v4f x3 = *(const v4f*)(ap + 12);
    v16bf a;
#pragma unroll
    for (int j = 0; j < 4; ++j) {
        a[j]      = (__bf16)x0[j];
        a[4 + j]  = (__bf16)x1[j];
        a[8 + j]  = (__bf16)x2[j];
        a[12 + j] = (__bf16)x3[j];
    }
    return a;
}

// ---------------------------------------------------------------------------
// Main fused kernel: per (b, 32-row T tile) workgroup, 8 waves.
// Waves split the U dimension: each wave owns 8 u-tiles, and carries TWO
// 16-row accumulator sets so each preloaded B fragment feeds 2 WMMAs:
//   per K-step: 8 B-fragment loads (one clause) + 2 A loads -> 16 WMMAs.
// Epilogue: score[b,t] += sum_u tanh(acc + qb[b,u]) * Wv[u]  (atomic f32).
// ---------------------------------------------------------------------------
__global__ void
__launch_bounds__(256)
attn_score_wmma_kernel(const float* __restrict__ values,
                       const __bf16* __restrict__ w2t,
                       const float* __restrict__ qb,
                       const float* __restrict__ Wv,
                       float* __restrict__ score) {
    const int b    = blockIdx.x >> 6;           // / (T_/32)
    const int t0   = (blockIdx.x & 63) << 5;    // 32 rows per workgroup
    const int lane = threadIdx.x & 31;
    const int wave = threadIdx.x >> 5;
    const int m    = lane & 15;                 // row (A) / column (B,C) index
    const int hi   = lane >> 4;                 // half-wave select

    const float* arow0 =
        values + ((size_t)(b * T_ + t0 + m)) * D_ + (hi << 4);
    const float* arow1 = arow0 + (size_t)16 * D_;

    v8f acc[2][8];
#pragma unroll
    for (int tf = 0; tf < 2; ++tf)
#pragma unroll
        for (int i = 0; i < 8; ++i) acc[tf][i] = (v8f)0.0f;

    // per-lane base for B fragments: element e <-> K = hi*16 + e (contiguous
    // order, as A); column u = (wave + 8i)*16 + m, K-contiguous in W2T.
    const __bf16* bbase = w2t + (size_t)((wave << 4) + m) * D_ + (hi << 4);

    for (int k0 = 0; k0 < D_; k0 += 32) {
        // preload all 8 B fragments (16 x b128 in one clause)
        v16bf bm[8];
#pragma unroll
        for (int i = 0; i < 8; ++i)
            bm[i] = *(const v16bf*)(bbase + (size_t)(i << 7) * D_ + k0);

        // two A fragments (shared across all 8 u-tiles each)
        v16bf a0 = load_a_frag(arow0 + k0);
        v16bf a1 = load_a_frag(arow1 + k0);

#pragma unroll
        for (int i = 0; i < 8; ++i)
            acc[0][i] = __builtin_amdgcn_wmma_f32_16x16x32_bf16(
                false, a0, false, bm[i], (short)0, acc[0][i], false, false);
#pragma unroll
        for (int i = 0; i < 8; ++i)
            acc[1][i] = __builtin_amdgcn_wmma_f32_16x16x32_bf16(
                false, a1, false, bm[i], (short)0, acc[1][i], false, false);
    }

    // Epilogue: h = tanh(vproj + qproj+b1+b2); partial score = h * Wv[u].
    // C layout: VGPR r, lane half hi -> row (t) = r + 8*hi, col (u) = m.
#pragma unroll
    for (int tf = 0; tf < 2; ++tf) {
#pragma unroll
        for (int i = 0; i < 8; ++i) {
            const int u     = (((i << 3) + wave) << 4) + m;
            const float wvv = Wv[u];
            const float qv  = qb[b * U_ + u];
            float p[8];
#pragma unroll
            for (int r = 0; r < 8; ++r)
                p[r] = tanhf(acc[tf][i][r] + qv) * wvv;
            // reduce across the 16 lanes of each half (over u columns)
#pragma unroll
            for (int s = 1; s < 16; s <<= 1) {
#pragma unroll
                for (int r = 0; r < 8; ++r)
                    p[r] += __shfl_xor(p[r], s, 32);
            }
            if (m == 0) {
#pragma unroll
                for (int r = 0; r < 8; ++r)
                    atomicAdd(&score[b * T_ + t0 + (tf << 4) + (hi << 3) + r],
                              p[r]);
            }
        }
    }
}

// ---------------------------------------------------------------------------
// Softmax over T per batch. (bv is a uniform shift -> drops out of softmax.)
// ---------------------------------------------------------------------------
__global__ void softmax_kernel(const float* __restrict__ score,
                               float* __restrict__ attn) {
    const int b = blockIdx.x;
    const int tid = threadIdx.x;
    __shared__ float red[256];

    float mx = -3.4e38f;
    for (int t = tid; t < T_; t += 256) mx = fmaxf(mx, score[b * T_ + t]);
    red[tid] = mx;
    __syncthreads();
    for (int s = 128; s > 0; s >>= 1) {
        if (tid < s) red[tid] = fmaxf(red[tid], red[tid + s]);
        __syncthreads();
    }
    mx = red[0];
    __syncthreads();

    float sum = 0.0f;
    for (int t = tid; t < T_; t += 256) sum += __expf(score[b * T_ + t] - mx);
    red[tid] = sum;
    __syncthreads();
    for (int s = 128; s > 0; s >>= 1) {
        if (tid < s) red[tid] += red[tid + s];
        __syncthreads();
    }
    const float inv = 1.0f / red[0];

    for (int t = tid; t < T_; t += 256)
        attn[b * T_ + t] = __expf(score[b * T_ + t] - mx) * inv;
}

// ---------------------------------------------------------------------------
// context[b,d] = sum_t attn[b,t] * values[b,t,d]   (coalesced over d)
// ---------------------------------------------------------------------------
__global__ void context_kernel(const float* __restrict__ values,
                               const float* __restrict__ attn,
                               float* __restrict__ ctx) {
    const int b = blockIdx.x >> 2;                    // D_/256 = 4 chunks
    const int d = ((blockIdx.x & 3) << 8) + threadIdx.x;
    const float* vb = values + (size_t)b * T_ * D_ + d;
    const float* ab = attn + b * T_;
    float acc = 0.0f;
    for (int t = 0; t < T_; ++t)
        acc += ab[t] * vb[(size_t)t * D_];
    ctx[b * D_ + d] = acc;
}

// ---------------------------------------------------------------------------
extern "C" void kernel_launch(void* const* d_in, const int* in_sizes, int n_in,
                              void* d_out, int out_size, void* d_ws, size_t ws_size,
                              hipStream_t stream) {
    const float* query  = (const float*)d_in[0];
    const float* values = (const float*)d_in[1];
    const float* W1     = (const float*)d_in[2];
    const float* b1     = (const float*)d_in[3];
    const float* W2     = (const float*)d_in[4];
    const float* b2     = (const float*)d_in[5];
    const float* Wv     = (const float*)d_in[6];
    // d_in[7] = bv: uniform additive shift, cancels in softmax; context uses
    // only attention weights -> bv has no effect on either output.

    float* ctx  = (float*)d_out;            // [B, D]
    float* attn = (float*)d_out + B_ * D_;  // [B, T, 1]

    // workspace: W2T bf16 (2 MB) | qb f32 (128 KB) | score f32 (256 KB)
    __bf16* w2t  = (__bf16*)d_ws;
    float*  qb   = (float*)((char*)d_ws + (size_t)U_ * D_ * sizeof(__bf16));
    float*  score = qb + B_ * U_;

    w2_transpose_bf16_kernel<<<(U_ * D_) / 256, 256, 0, stream>>>(W2, w2t);
    qproj_kernel<<<B_ * (U_ / 256), 256, 0, stream>>>(query, W1, b1, b2, qb);
    zero_kernel<<<(B_ * T_) / 256, 256, 0, stream>>>(score);
    attn_score_wmma_kernel<<<B_ * (T_ / 32), 256, 0, stream>>>(values, w2t, qb,
                                                               Wv, score);
    softmax_kernel<<<B_, 256, 0, stream>>>(score, attn);
    context_kernel<<<B_ * (D_ / 256), 256, 0, stream>>>(values, attn, ctx);
}
